// TtMultiHeadAttentionModel_29351806501057
// MI455X (gfx1250) — matrix operand order repaired and checked by reading, not compiled, not run on hardware
//
#include <hip/hip_runtime.h>

// ---------------------------------------------------------------------------
// Fused MHA for MI455X (gfx1250, wave32, WMMA bf16 16x16x32, f32 accumulate)
// B=16, S=384, H=1024, NH=16, HD=64
// ---------------------------------------------------------------------------

typedef __bf16 bf16;
typedef __attribute__((ext_vector_type(16))) __bf16       v16bf;
typedef __attribute__((ext_vector_type(8)))  float        v8f;
typedef __attribute__((ext_vector_type(4)))  __bf16       v4bf;
typedef __attribute__((ext_vector_type(4)))  unsigned int u32x4;

#define B_   16
#define S_   384
#define H_   1024
#define NH_  16
#define HD_  64
#define N3H_ 3072

union FragA { v16bf bf; u32x4 u[2]; };

__device__ __forceinline__ v8f wmma_bf16(v16bf a, v16bf b, v8f c) {
  // D = A(16x32) * B(32x16) + C, f32 accumulate
  return __builtin_amdgcn_wmma_f32_16x16x32_bf16(false, a, false, b, (short)0, c,
                                                 false, false);
}

// ---------------------------------------------------------------------------
// Kernel 1: activation f32 -> bf16  (6144*1024 elems, float4 vectorized)
// ---------------------------------------------------------------------------
__global__ void convert_x_kernel(const float* __restrict__ x, bf16* __restrict__ xb) {
  int i = blockIdx.x * blockDim.x + threadIdx.x;   // one float4 per thread
  float4 v = ((const float4*)x)[i];
  v4bf r;
  r[0] = (bf16)v.x; r[1] = (bf16)v.y; r[2] = (bf16)v.z; r[3] = (bf16)v.w;
  ((v4bf*)xb)[i] = r;
}

// ---------------------------------------------------------------------------
// Kernel 2: qkv_weight f32 [K=1024, N=3072] -> bf16 transposed Wt [N, K]
// 32x32 LDS tile transpose, coalesced read & write.
// ---------------------------------------------------------------------------
__global__ void convert_w_kernel(const float* __restrict__ w, bf16* __restrict__ wt) {
  __shared__ float tile[32][33];
  int nb = blockIdx.x * 32;   // output-feature base (0..3071)
  int kb = blockIdx.y * 32;   // K base (0..1023)
  int t  = threadIdx.x;
#pragma unroll
  for (int i = 0; i < 4; ++i) {
    int idx = t + 256 * i;
    int r = idx >> 5, c = idx & 31;                 // r = k, c = n
    tile[r][c] = w[(size_t)(kb + r) * N3H_ + nb + c];
  }
  __syncthreads();
#pragma unroll
  for (int i = 0; i < 4; ++i) {
    int idx = t + 256 * i;
    int r = idx >> 5, c = idx & 31;                 // r = n, c = k
    wt[(size_t)(nb + r) * H_ + kb + c] = (bf16)tile[c][r];
  }
}

// ---------------------------------------------------------------------------
// Kernel 3: QKV GEMM  C[m,n] = sum_k Xb[m,k] * Wt[n,k] + bias[n]
// M=6144 N=3072 K=1024.  Block tile 128x128x32, 8 waves (2x4), wave tile 64x32.
// Epilogue scatters into Q[b,h,s,d], K[b,h,s,d], Vt[b,h,d,s] as bf16.
// ---------------------------------------------------------------------------
__global__ void __launch_bounds__(256)
gemm_qkv_kernel(const bf16* __restrict__ Xb, const bf16* __restrict__ Wt,
                const float* __restrict__ bias,
                bf16* __restrict__ Qw, bf16* __restrict__ Kw, bf16* __restrict__ Vtw) {
  __shared__ bf16 ldsA[2][128][32];
  __shared__ bf16 ldsB[2][128][32];

  const int t    = threadIdx.x;
  const int wave = t >> 5, lane = t & 31;
  const int l16  = lane & 15, half = lane >> 4;
  const int wm   = wave >> 2, wn = wave & 3;       // 2 x 4 wave grid
  const int mblock = blockIdx.y * 128;
  const int nblock = blockIdx.x * 128;

  // staging: each thread copies 32B of A and 32B of B per buffer
  const int srow = t >> 1, skoff = (t & 1) * 16;

  auto stage = [&](int buf, int kbase) {
    const bf16* ap = Xb + (size_t)(mblock + srow) * H_ + kbase + skoff;
    *(u32x4*)&ldsA[buf][srow][skoff]     = *(const u32x4*)(ap);
    *(u32x4*)&ldsA[buf][srow][skoff + 8] = *(const u32x4*)(ap + 8);
    const bf16* bp = Wt + (size_t)(nblock + srow) * H_ + kbase + skoff;
    *(u32x4*)&ldsB[buf][srow][skoff]     = *(const u32x4*)(bp);
    *(u32x4*)&ldsB[buf][srow][skoff + 8] = *(const u32x4*)(bp + 8);
  };

  v8f acc[4][2] = {};

  stage(0, 0);
  __syncthreads();

  for (int kk = 0; kk < H_ / 32; ++kk) {
    const int buf = kk & 1;
    if (kk + 1 < H_ / 32) stage(buf ^ 1, (kk + 1) * 32);

    FragA af[4], bfg[2];
#pragma unroll
    for (int mt = 0; mt < 4; ++mt) {
      const bf16* p = &ldsA[buf][wm * 64 + mt * 16 + l16][0];
      af[mt].u[0] = *(const u32x4*)(p + 8 * half);         // K {0..7}/{8..15}
      af[mt].u[1] = *(const u32x4*)(p + 16 + 8 * half);    // K {16..23}/{24..31}
    }
#pragma unroll
    for (int nt = 0; nt < 2; ++nt) {
      const bf16* p = &ldsB[buf][wn * 32 + nt * 16 + l16][0];
      bfg[nt].u[0] = *(const u32x4*)(p + 8 * half);        // K {0..7}/{8..15}
      bfg[nt].u[1] = *(const u32x4*)(p + 16 + 8 * half);
    }
#pragma unroll
    for (int mt = 0; mt < 4; ++mt)
#pragma unroll
      for (int nt = 0; nt < 2; ++nt)
        acc[mt][nt] = wmma_bf16(af[mt].bf, bfg[nt].bf, acc[mt][nt]);

    __syncthreads();
  }

  // epilogue: bias add + scatter to per-head Q/K/Vt (bf16)
  float bv[2];
  bv[0] = bias[nblock + wn * 32 + l16];
  bv[1] = bias[nblock + wn * 32 + 16 + l16];

#pragma unroll
  for (int mt = 0; mt < 4; ++mt) {
#pragma unroll
    for (int nt = 0; nt < 2; ++nt) {
      const int n = nblock + wn * 32 + nt * 16 + l16;
      const int which = n >> 10;                 // 0=Q 1=K 2=V (uniform per tile)
      const int f = n & 1023, h = f >> 6, d = f & 63;
#pragma unroll
      for (int r = 0; r < 8; ++r) {
        const int m = mblock + wm * 64 + mt * 16 + r + 8 * half;
        const int b = m / S_, s = m - b * S_;
        const float val = acc[mt][nt][r] + bv[nt];
        const size_t bh = (size_t)(b * NH_ + h);
        if (which == 0)      Qw[(bh * S_ + s) * HD_ + d]  = (bf16)val;
        else if (which == 1) Kw[(bh * S_ + s) * HD_ + d]  = (bf16)val;
        else                 Vtw[(bh * HD_ + d) * S_ + s] = (bf16)val;
      }
    }
  }
}

// ---------------------------------------------------------------------------
// Kernel 4: flash attention.  Block = (b, h, 128 q rows); 8 waves x 16 q rows.
// Computes s^T = K_j * Q^T (so softmax reduction is mostly in-lane), then
// o^T += V^T * P^T.  K/Vt key-tiles (32 keys) double-buffered in LDS.
// ---------------------------------------------------------------------------
__global__ void __launch_bounds__(256)
attention_kernel(const bf16* __restrict__ Qw, const bf16* __restrict__ Kw,
                 const bf16* __restrict__ Vtw, const float* __restrict__ mask,
                 float* __restrict__ out) {
  __shared__ bf16  ldsK[2][32][64];   // [key][d]
  __shared__ bf16  ldsV[2][64][32];   // [d][key]   (Vt tile)
  __shared__ float ldsM[S_];

  const int t    = threadIdx.x;
  const int wave = t >> 5, lane = t & 31;
  const int l16  = lane & 15, half = lane >> 4;
  const int qb   = blockIdx.x % (S_ / 128);
  const int bh   = blockIdx.x / (S_ / 128);
  const int b    = bh >> 4;

  const bf16* Qp = Qw  + (size_t)bh * S_ * HD_;
  const bf16* Kp = Kw  + (size_t)bh * S_ * HD_;
  const bf16* Vp = Vtw + (size_t)bh * HD_ * S_;

  for (int i = t; i < S_; i += 256) ldsM[i] = mask[b * S_ + i];

  // Q as B-fragment (Q^T columns): lane q = l16, d = 32*s2 + {8*half + 0..7, 16 + 8*half + 0..7}
  const int qr = qb * 128 + wave * 16 + l16;
  FragA qf[2];
#pragma unroll
  for (int s2 = 0; s2 < 2; ++s2) {
    const bf16* p = Qp + (size_t)qr * HD_ + 32 * s2;
    qf[s2].u[0] = *(const u32x4*)(p + 8 * half);
    qf[s2].u[1] = *(const u32x4*)(p + 16 + 8 * half);
  }

  // cooperative staging of K tile (32x64) and Vt tile (64x32)
  const int krow = t >> 3, kdo = (t & 7) * 8;
  const int vrow = t >> 2, vko = (t & 3) * 8;
  auto stage = [&](int buf, int jb) {
    *(u32x4*)&ldsK[buf][krow][kdo] =
        *(const u32x4*)(Kp + (size_t)(jb + krow) * HD_ + kdo);
    *(u32x4*)&ldsV[buf][vrow][vko] =
        *(const u32x4*)(Vp + (size_t)vrow * S_ + jb + vko);
  };

  v8f   acc[4] = {};
  float m_s = -1e30f, l_s = 0.f;
  const float scale = 0.125f;       // 1/sqrt(64)

  stage(0, 0);
  __syncthreads();

  for (int j = 0; j < S_ / 32; ++j) {
    const int buf = j & 1;
    if (j + 1 < S_ / 32) stage(buf ^ 1, (j + 1) * 32);
    const int jb = j * 32;

    // --- s^T tiles: st0 = keys 0..15, st1 = keys 16..31 (x q 16) ---
    v8f st0 = {}, st1 = {};
#pragma unroll
    for (int s2 = 0; s2 < 2; ++s2) {
      FragA kf0, kf1;
      const bf16* p0 = &ldsK[buf][l16][32 * s2];
      const bf16* p1 = &ldsK[buf][16 + l16][32 * s2];
      kf0.u[0] = *(const u32x4*)(p0 + 8 * half);
      kf0.u[1] = *(const u32x4*)(p0 + 16 + 8 * half);
      kf1.u[0] = *(const u32x4*)(p1 + 8 * half);
      kf1.u[1] = *(const u32x4*)(p1 + 16 + 8 * half);
      st0 = wmma_bf16(kf0.bf, qf[s2].bf, st0);
      st1 = wmma_bf16(kf1.bf, qf[s2].bf, st1);
    }

    // --- online softmax (per lane: 16 keys for its q; pair covers 32) ---
    float e0[8], e1[8], bm = -1e30f;
#pragma unroll
    for (int r = 0; r < 8; ++r) {
      float x = st0[r] * scale + ldsM[jb + r + 8 * half];
      e0[r] = x; bm = fmaxf(bm, x);
    }
#pragma unroll
    for (int r = 0; r < 8; ++r) {
      float x = st1[r] * scale + ldsM[jb + 16 + r + 8 * half];
      e1[r] = x; bm = fmaxf(bm, x);
    }
    bm = fmaxf(bm, __shfl_xor(bm, 16));
    const float mnew  = fmaxf(m_s, bm);
    const float alpha = __expf(m_s - mnew);
    float lsum = 0.f;
#pragma unroll
    for (int r = 0; r < 8; ++r) { e0[r] = __expf(e0[r] - mnew); lsum += e0[r]; }
#pragma unroll
    for (int r = 0; r < 8; ++r) { e1[r] = __expf(e1[r] - mnew); lsum += e1[r]; }
    lsum += __shfl_xor(lsum, 16);
    l_s = l_s * alpha + lsum;
    m_s = mnew;
#pragma unroll
    for (int mt = 0; mt < 4; ++mt)
#pragma unroll
      for (int r = 0; r < 8; ++r) acc[mt][r] *= alpha;

    // --- build P^T B-fragment via half-wave exchange (keys -> packed K) ---
    float p0[8], p1[8];
#pragma unroll
    for (int r = 0; r < 8; ++r) p0[r] = __shfl_xor(e0[r], 16);
#pragma unroll
    for (int r = 0; r < 8; ++r) p1[r] = __shfl_xor(e1[r], 16);
    FragA pf;
#pragma unroll
    for (int r = 0; r < 8; ++r) {
      float lo = (half == 0) ? e0[r] : p1[r];   // frag element r
      float hi = (half == 0) ? p0[r] : e1[r];   // frag element r+8
      pf.bf[r]     = (bf16)lo;
      pf.bf[r + 8] = (bf16)hi;
    }

    // --- o^T += V^T(16d x 32keys) * P^T(32keys x 16q), 4 d-tiles ---
#pragma unroll
    for (int mt = 0; mt < 4; ++mt) {
      FragA vf;
      const bf16* p = &ldsV[buf][mt * 16 + l16][0];
      vf.u[0] = *(const u32x4*)(p + 16 * half);
      vf.u[1] = *(const u32x4*)(p + 16 * half + 8);
      acc[mt] = wmma_bf16(vf.bf, pf.bf, acc[mt]);
    }
    __syncthreads();
  }

  // --- epilogue: normalize, write [B,S,H] f32 (concat heads by indexing) ---
  const float invl = 1.f / l_s;
  const int h = bh & 15;
  const size_t obase = ((size_t)b * S_ + qr) * H_ + h * HD_;
#pragma unroll
  for (int mt = 0; mt < 4; ++mt)
#pragma unroll
    for (int r = 0; r < 8; ++r) {
      const int d = mt * 16 + r + 8 * half;
      out[obase + d] = acc[mt][r] * invl;
    }
}

// ---------------------------------------------------------------------------
// launch
// ---------------------------------------------------------------------------
extern "C" void kernel_launch(void* const* d_in, const int* in_sizes, int n_in,
                              void* d_out, int out_size, void* d_ws, size_t ws_size,
                              hipStream_t stream) {
  const float* act  = (const float*)d_in[0];   // [B,S,H]
  const float* mask = (const float*)d_in[1];   // [B,1,1,S]
  const float* w    = (const float*)d_in[2];   // [H,3H]
  const float* bias = (const float*)d_in[3];   // [3H]
  float*       out  = (float*)d_out;           // [B,S,H]

  char* ws = (char*)d_ws;
  const size_t xb_bytes  = (size_t)B_ * S_ * H_ * sizeof(bf16);       // 12.6 MB
  const size_t wt_bytes  = (size_t)N3H_ * H_ * sizeof(bf16);          //  6.3 MB
  const size_t hd_bytes  = (size_t)B_ * NH_ * S_ * HD_ * sizeof(bf16);// 12.6 MB each
  bf16* Xb  = (bf16*)(ws);
  bf16* Wt  = (bf16*)(ws + xb_bytes);
  bf16* Qw  = (bf16*)(ws + xb_bytes + wt_bytes);
  bf16* Kw  = (bf16*)(ws + xb_bytes + wt_bytes + hd_bytes);
  bf16* Vtw = (bf16*)(ws + xb_bytes + wt_bytes + 2 * hd_bytes);

  // 1) activation -> bf16
  convert_x_kernel<<<(B_ * S_ * H_) / 4 / 256, 256, 0, stream>>>(act, Xb);
  // 2) weight -> bf16 transposed
  convert_w_kernel<<<dim3(N3H_ / 32, H_ / 32), 256, 0, stream>>>(w, Wt);
  // 3) QKV projection GEMM (writes per-head Q, K, Vt)
  gemm_qkv_kernel<<<dim3(N3H_ / 128, (B_ * S_) / 128), 256, 0, stream>>>(
      Xb, Wt, bias, Qw, Kw, Vtw);
  // 4) flash attention
  attention_kernel<<<B_ * NH_ * (S_ / 128), 256, 0, stream>>>(Qw, Kw, Vtw, mask, out);
}
